// CapsuleEncoder_4612794876421
// MI455X (gfx1250) — matrix operand order repaired and checked by reading, not compiled, not run on hardware
//
#include <hip/hip_runtime.h>
#include <hip/hip_bf16.h>
#include <math.h>

typedef __attribute__((ext_vector_type(16))) _Float16 v16h;
typedef __attribute__((ext_vector_type(8)))  _Float16 v8h;
typedef __attribute__((ext_vector_type(8)))  float    v8f;
typedef __attribute__((ext_vector_type(2)))  float    v2f;

#define BATCH   16
#define H0      47
#define H1      39          // 47-9+1
#define H2      16          // (39-9)/2+1
#define C1      128
#define MDIM    1024        // prim out channels
#define KDIM    10368       // 128*81
#define NDIM    4096        // 16*16*16 (b,oy,ox)
#define CIN     16384       // 1024*16
#define DDIM    16
#define KSPLIT  16          // caps einsum split-K factor
#define KCHUNK  (CIN / KSPLIT)  // 1024

#define BM      128
#define BN      128
#define BK      32
#define KSTEPS  (KDIM / BK) // 324

// ---------------------------------------------------------------------------
// conv1 + bias + ReLU, fp32 in, f16 out (1.5 GFLOP; VALU)
// ---------------------------------------------------------------------------
__global__ void conv1_relu_f16(const float* __restrict__ x, const float* __restrict__ w,
                               const float* __restrict__ bias, _Float16* __restrict__ y) {
  int idx = blockIdx.x * blockDim.x + threadIdx.x;
  const int total = BATCH * C1 * H1 * H1;
  if (idx >= total) return;
  int ox = idx % H1; int t = idx / H1;
  int oy = t % H1;   t /= H1;
  int oc = t % C1;   int b = t / C1;
  float acc = bias[oc];
  for (int c = 0; c < 3; ++c) {
    const float* xp = x + ((b * 3 + c) * H0 + oy) * H0 + ox;
    const float* wp = w + (oc * 3 + c) * 81;
#pragma unroll
    for (int ky = 0; ky < 9; ++ky)
#pragma unroll
      for (int kx = 0; kx < 9; ++kx)
        acc = fmaf(xp[ky * H0 + kx], wp[ky * 9 + kx], acc);
  }
  y[idx] = (_Float16)fmaxf(acc, 0.f);
}

// ---------------------------------------------------------------------------
__global__ void cvt_f32_f16(const float* __restrict__ src, _Float16* __restrict__ dst, int n) {
  int i = blockIdx.x * blockDim.x + threadIdx.x;
  if (i < n) dst[i] = (_Float16)src[i];
}

// ---------------------------------------------------------------------------
// im2col; P stored N-major: P[n][k], n=b*256+oy*16+ox  (B-fragment loads contiguous)
// ---------------------------------------------------------------------------
__global__ void im2col_f16(const _Float16* __restrict__ y, _Float16* __restrict__ P) {
  int k = blockIdx.x * blockDim.x + threadIdx.x;
  if (k >= KDIM) return;
  int n = blockIdx.y;
  int c = k / 81, r = k % 81, ky = r / 9, kx = r % 9;
  int b = n >> 8, oy = (n >> 4) & 15, ox = n & 15;
  P[(size_t)n * KDIM + k] = y[((b * C1 + c) * H1 + 2 * oy + ky) * H1 + 2 * ox + kx];
}

// ---------------------------------------------------------------------------
// CDNA5 async copy: 16B per lane, global -> LDS, tracked by ASYNCcnt
// ---------------------------------------------------------------------------
__device__ __forceinline__ void async_cp16(unsigned lds_off, const void* gptr) {
  unsigned long long ga = (unsigned long long)(uintptr_t)gptr;
  asm volatile("global_load_async_to_lds_b128 %0, %1, off" :: "v"(lds_off), "v"(ga) : "memory");
}

// ---------------------------------------------------------------------------
// WMMA fragment loaders from LDS (layouts per CDNA5 ISA §7.12.2)
// ---------------------------------------------------------------------------
__device__ __forceinline__ v16h ldA16(const _Float16* p) {   // p = rowbase + (lane<16?0:8)
  v8h lo = *(const v8h*)p;
  v8h hi = *(const v8h*)(p + 16);
  v16h r;
#pragma unroll
  for (int i = 0; i < 8; ++i) { r[i] = lo[i]; r[8 + i] = hi[i]; }
  return r;
}
__device__ __forceinline__ v16h ldB16(const _Float16* p) {   // p = colbase + (lane<16?0:8)
  v8h lo = *(const v8h*)p;
  v8h hi = *(const v8h*)(p + 16);
  v16h r;
#pragma unroll
  for (int i = 0; i < 8; ++i) { r[i] = lo[i]; r[8 + i] = hi[i]; }
  return r;
}

// ---------------------------------------------------------------------------
// Primary-caps GEMM: C[1024][4096] = A[1024][10368](f16) x P^T, 128x128 LDS-blocked,
// double-buffered CDNA5 async global->LDS copies, 8 waves x 8 WMMA per k-step.
// ---------------------------------------------------------------------------
__global__ void __launch_bounds__(256) gemm_prim(const _Float16* __restrict__ A,
                                                 const _Float16* __restrict__ P,
                                                 float* __restrict__ C) {
  __shared__ _Float16 As[2][BM][BK];   // 2 x 8KB
  __shared__ _Float16 Bs[2][BN][BK];   // 2 x 8KB
  const int tid  = threadIdx.x;
  const int lane = tid & 31;
  const int wid  = tid >> 5;           // 0..7
  const int bm = blockIdx.x >> 5, bn = blockIdx.x & 31;
  const int m0 = bm * BM, n0 = bn * BN;

  // ---- async-loader mapping: thread covers rows r and r+64, 16B quarter q ----
  const int r = tid >> 2;              // 0..63
  const int q = tid & 3;               // 16B quarter within the 64B row
  const _Float16* gA0 = A + (size_t)(m0 + r) * KDIM + q * 8;
  const _Float16* gA1 = gA0 + (size_t)64 * KDIM;
  const _Float16* gB0 = P + (size_t)(n0 + r) * KDIM + q * 8;
  const _Float16* gB1 = gB0 + (size_t)64 * KDIM;
  const unsigned aBase = (unsigned)(uintptr_t)(&As[0][0][0]);
  const unsigned bBase = (unsigned)(uintptr_t)(&Bs[0][0][0]);
  const unsigned lA0 = aBase + (unsigned)(r * BK + q * 8) * 2;
  const unsigned lA1 = lA0 + 64 * BK * 2;
  const unsigned lB0 = bBase + (unsigned)(r * BK + q * 8) * 2;
  const unsigned lB1 = lB0 + 64 * BK * 2;

  // ---- compute mapping: wave (wm,wn) owns a 64x32 C sub-tile ----
  const int wm = wid >> 2, wn = wid & 3;
  const int row = lane & 15;
  const int kbA = (lane < 16) ? 0 : 8;
  const int kbB = (lane < 16) ? 0 : 8;
  v8f acc[4][2];
#pragma unroll
  for (int mt = 0; mt < 4; ++mt)
#pragma unroll
    for (int nt = 0; nt < 2; ++nt) acc[mt][nt] = (v8f){};

  auto issue = [&](int buf, int kk) {
    unsigned bo = (unsigned)buf * (BM * BK * 2);
    async_cp16(lA0 + bo, gA0 + kk);
    async_cp16(lA1 + bo, gA1 + kk);
    async_cp16(lB0 + bo, gB0 + kk);
    async_cp16(lB1 + bo, gB1 + kk);
  };
  auto compute = [&](int buf) {
    const _Float16* Ab = &As[buf][wm * 64][0];
    const _Float16* Bb = &Bs[buf][wn * 32][0];
    v16h af[4], bf[2];
#pragma unroll
    for (int mt = 0; mt < 4; ++mt) af[mt] = ldA16(Ab + (mt * 16 + row) * BK + kbA);
#pragma unroll
    for (int nt = 0; nt < 2; ++nt) bf[nt] = ldB16(Bb + (nt * 16 + row) * BK + kbB);
#pragma unroll
    for (int mt = 0; mt < 4; ++mt)
#pragma unroll
      for (int nt = 0; nt < 2; ++nt)
        acc[mt][nt] = __builtin_amdgcn_wmma_f32_16x16x32_f16(
            false, af[mt], false, bf[nt], (short)0, acc[mt][nt], false, false);
  };

  issue(0, 0);
  for (int it = 0; it < KSTEPS - 1; ++it) {
    int buf = it & 1;
    issue(buf ^ 1, (it + 1) * BK);
    asm volatile("s_wait_asynccnt 4" ::: "memory");   // current tile's 4 copies done
    __syncthreads();                                   // visible to all waves
    compute(buf);
    __syncthreads();                                   // done reading before overwrite
  }
  asm volatile("s_wait_asynccnt 0" ::: "memory");
  __syncthreads();
  compute((KSTEPS - 1) & 1);

  const int col = lane & 15, mb = (lane < 16) ? 0 : 8;
#pragma unroll
  for (int mt = 0; mt < 4; ++mt)
#pragma unroll
    for (int nt = 0; nt < 2; ++nt) {
      float* cp = C + (size_t)(m0 + wm * 64 + mt * 16 + mb) * NDIM + n0 + wn * 32 + nt * 16 + col;
#pragma unroll
      for (int rr = 0; rr < 8; ++rr) cp[(size_t)rr * NDIM] = acc[mt][nt][rr];
    }
}

// ---------------------------------------------------------------------------
// bias + squash over the 16 ox values; scatter into Xt[b][w=ox][c=ch*16+oy]
// ---------------------------------------------------------------------------
__global__ void bias_squash_xt(const float* __restrict__ C, const float* __restrict__ bias,
                               float* __restrict__ Xt) {
  int tid = blockIdx.x * blockDim.x + threadIdx.x;   // 262144
  int m = tid >> 8, vi = tid & 255;
  const float* cp = C + (size_t)m * NDIM + vi * 16;
  float bq = bias[m], v[16], sq = 0.f;
#pragma unroll
  for (int j = 0; j < 16; ++j) { v[j] = cp[j] + bq; sq = fmaf(v[j], v[j], sq); }
  float scale = sqrtf(sq) / (1.f + sq);
  int b = vi >> 4, oy = vi & 15;
  float* xp = Xt + (size_t)b * 16 * CIN + m * 16 + oy;
#pragma unroll
  for (int ox = 0; ox < 16; ++ox) xp[(size_t)ox * CIN] = v[ox] * scale;
}

// ---------------------------------------------------------------------------
// caps einsum, split-K x16: 4096 waves, each a (16 x 1024)*(1024 x 16) partial GEMM
// with v_wmma_f32_16x16x4_f32. caps_w (256 MB fp32) read exactly once; 16x the
// memory-level parallelism of the unsplit version -> can approach 23.3 TB/s.
// ---------------------------------------------------------------------------
__global__ void caps_gemm_split(const float* __restrict__ Xt, const float* __restrict__ Wc,
                                float* __restrict__ Tp) {
  int w = (blockIdx.x * blockDim.x + threadIdx.x) >> 5;  // 0..4095
  int lane = threadIdx.x & 31;
  int pair = w >> 4, chunk = w & 15;
  int b = pair >> 4, n = pair & 15;
  int row  = lane & 15;
  int koff = (lane < 16) ? 0 : 2;
  int k0 = chunk * KCHUNK;
  const float* ap = Xt + (size_t)b * 16 * CIN + (size_t)row * CIN + k0 + koff;
  const float* bp = Wc + (size_t)(b * 16 + n) * CIN * DDIM + (size_t)(k0 + koff) * DDIM + row;
  v8f acc = {};
#pragma unroll 8
  for (int k = 0; k < KCHUNK; k += 4) {
    v2f a = *(const v2f*)(ap + k);
    v2f bb;
    bb[0] = bp[(size_t)k * DDIM];
    bb[1] = bp[(size_t)k * DDIM + DDIM];
    acc = __builtin_amdgcn_wmma_f32_16x16x4_f32(false, a, false, bb, (short)0, acc, false, false);
  }
  int mb = (lane < 16) ? 0 : 8;
  float* tp = Tp + (size_t)w * 256;                   // partial tile, same layout as T
#pragma unroll
  for (int rr = 0; rr < 8; ++rr) tp[(mb + rr) * 16 + row] = acc[rr];
}

__global__ void reduce_t(const float* __restrict__ Tp, float* __restrict__ T) {
  int idx = blockIdx.x * blockDim.x + threadIdx.x;     // 65536
  int pair = idx >> 8, e = idx & 255;
  float s = 0.f;
#pragma unroll
  for (int c = 0; c < KSPLIT; ++c) s += Tp[((size_t)(pair * KSPLIT + c)) * 256 + e];
  T[idx] = s;
}

// ---------------------------------------------------------------------------
// Dynamic routing (3 iters), faithful to the reference broadcast:
// logits[b,k,n,d] += s[k,n,d]
// ---------------------------------------------------------------------------
__global__ void zero_f32(float* p, int n) {
  int i = blockIdx.x * blockDim.x + threadIdx.x;
  if (i < n) p[i] = 0.f;
}

__global__ void routing_step(const float* __restrict__ T, const float* __restrict__ logits,
                             float* __restrict__ outv, float* __restrict__ S, int compute_s) {
  int p = blockIdx.x * blockDim.x + threadIdx.x;      // 0..255 (b*16+k)
  const float* t  = T + (size_t)p * 256;
  const float* lg = logits + (size_t)p * 256;
  float out[16], sq = 0.f;
#pragma unroll
  for (int d = 0; d < 16; ++d) {
    float m = -3.4e38f;
    for (int nn = 0; nn < 16; ++nn) m = fmaxf(m, lg[nn * 16 + d]);
    float den = 0.f, num = 0.f;
    for (int nn = 0; nn < 16; ++nn) {
      float e = expf(lg[nn * 16 + d] - m);
      den += e;
      num = fmaf(e, t[nn * 16 + d], num);
    }
    out[d] = num / den;
  }
#pragma unroll
  for (int d = 0; d < 16; ++d) sq = fmaf(out[d], out[d], sq);
  float scale = sqrtf(sq) / (1.f + sq);
#pragma unroll
  for (int d = 0; d < 16; ++d) { out[d] *= scale; outv[p * 16 + d] = out[d]; }
  if (compute_s) {
    for (int nn = 0; nn < 16; ++nn) {
      float s = 0.f;
      for (int d = 0; d < 16; ++d) s = fmaf(t[nn * 16 + d], out[d], s);
      S[p * 16 + nn] = s;
    }
  }
}

__global__ void routing_update(float* __restrict__ logits, const float* __restrict__ S) {
  int idx = blockIdx.x * blockDim.x + threadIdx.x;     // 65536
  if (idx < BATCH * 16 * 16 * 16)
    logits[idx] += S[idx & 4095];                      // S[k*256+n*16+d]
}

// ---------------------------------------------------------------------------
extern "C" void kernel_launch(void* const* d_in, const int* in_sizes, int n_in,
                              void* d_out, int out_size, void* d_ws, size_t ws_size,
                              hipStream_t stream) {
  const float* x      = (const float*)d_in[0];
  const float* conv_w = (const float*)d_in[1];
  const float* conv_b = (const float*)d_in[2];
  const float* prim_w = (const float*)d_in[3];
  const float* prim_b = (const float*)d_in[4];
  const float* caps_w = (const float*)d_in[5];
  float* out = (float*)d_out;

  size_t off = 0;
  auto carve = [&](size_t bytes) { size_t r = off; off += (bytes + 255) & ~(size_t)255; return r; };
  char* ws = (char*)d_ws;
  const size_t Y_ELEMS = (size_t)BATCH * C1 * H1 * H1;       // 3,115,008
  const size_t W_ELEMS = (size_t)MDIM * KDIM;                // 10,616,832
  _Float16* yf16 = (_Float16*)(ws + carve(Y_ELEMS * 2));
  _Float16* wf16 = (_Float16*)(ws + carve(W_ELEMS * 2));
  _Float16* P    = (_Float16*)(ws + carve((size_t)NDIM * KDIM * 2));
  float*    Cbuf = (float*)(ws + carve((size_t)MDIM * NDIM * 4));
  float*    Xt   = (float*)(ws + carve((size_t)BATCH * 16 * CIN * 4));
  float*    Tpar = (float*)(ws + carve((size_t)4096 * 256 * 4));   // split-K partials (4MB)
  float*    T    = (float*)(ws + carve((size_t)256 * 256 * 4));
  float*    LG   = (float*)(ws + carve((size_t)256 * 256 * 4));
  float*    OV   = (float*)(ws + carve((size_t)256 * 16 * 4));
  float*    S    = (float*)(ws + carve((size_t)256 * 16 * 4));

  conv1_relu_f16<<<(int)((Y_ELEMS + 255) / 256), 256, 0, stream>>>(x, conv_w, conv_b, yf16);
  cvt_f32_f16<<<(int)((W_ELEMS + 255) / 256), 256, 0, stream>>>(prim_w, wf16, (int)W_ELEMS);
  im2col_f16<<<dim3((KDIM + 255) / 256, NDIM), 256, 0, stream>>>(yf16, P);
  gemm_prim<<<256, 256, 0, stream>>>(wf16, P, Cbuf);           // 8x32 blocks of 128x128
  bias_squash_xt<<<1024, 256, 0, stream>>>(Cbuf, prim_b, Xt);
  caps_gemm_split<<<1024, 128, 0, stream>>>(Xt, caps_w, Tpar); // 4096 waves
  reduce_t<<<256, 256, 0, stream>>>(Tpar, T);
  zero_f32<<<256, 256, 0, stream>>>(LG, 256 * 256);
  for (int it = 0; it < 3; ++it) {
    float* dst = (it == 2) ? out : OV;
    routing_step<<<8, 32, 0, stream>>>(T, LG, dst, S, (it < 2) ? 1 : 0);
    if (it < 2) routing_update<<<256, 256, 0, stream>>>(LG, S);
  }
}
